// Network_21912923144546
// MI455X (gfx1250) — matrix operand order repaired and checked by reading, not compiled, not run on hardware
//
#include <hip/hip_runtime.h>
#include <math.h>

// ---------------- types ----------------
typedef __bf16 bf16_t;
typedef __attribute__((ext_vector_type(16))) __bf16 v16bf;
typedef __attribute__((ext_vector_type(8)))  __bf16 v8bf;
typedef __attribute__((ext_vector_type(8)))  float  v8f;

#define NB   2
#define NT   2048
#define DFEAT 1024
#define NDIM 512
#define NH   8
#define NDH  64
#define NINNER 512
#define NMLP 512
#define NRED1 32
#define NRED2 16
#define EPSF 1e-5f
#define INV_E 0.36787944117144233f

enum { ACT_NONE = 0, ACT_RELU = 1, ACT_GELU = 2 };

// gfx1250 async global->LDS copy (16B), tracked by ASYNCcnt
__device__ __forceinline__ void async_ld_b128(unsigned lds_off, const bf16_t* g)
{
    asm volatile("global_load_async_to_lds_b128 %0, %1, off"
                 :: "v"(lds_off), "v"(g) : "memory");
}
__device__ __forceinline__ void wait_async0()
{
    asm volatile("s_wait_asynccnt 0" ::: "memory");
}

// =====================================================================
// bf16 WMMA GEMM:  C[M,N] = act(A[M,K] @ Bt[N,K]^T + bias) + residual
//  - A row-major [M,K]; B ALWAYS stored transposed, row-major [N,K].
//  - Block: 256 thr = 8 wave32 waves; block tile 128x64; K step 32.
//  - Each wave: 32x32 strip -> 4 x v_wmma_f32_16x16x32_bf16 per K step.
//  - LDS holds fragments in WMMA lane order (ISA 7.12.2), double buffered,
//    filled with global_load_async_to_lds_b128 (16B global -> 16B LDS).
// Fragment-order mapping (16-bit 16x32 A tile, 512 elements):
//   lane holds 16 contiguous bf16: elems[0..7] = K group g0, [8..15] = g1
//   A row r, K chunk c(8): lane=(r&15)+16*((c>>3)&1), eoff=(c&16)?8:0
//   Bt row n, K chunk c(8): lane=(n&15)+16*((c>>3)&1), eoff=(c&16)?8:0
// =====================================================================
__global__ __launch_bounds__(256)
void gemm_bf16_wmma(const bf16_t* __restrict__ A, int lda,
                    const bf16_t* __restrict__ Bt, int ldb,
                    float* __restrict__ Cf, bf16_t* __restrict__ Cb,
                    bf16_t* __restrict__ CbT, int ldc, int ldcT,
                    const float* __restrict__ bias,
                    const float* __restrict__ residual, int ldres,
                    int K, int act)
{
    __shared__ __align__(16) bf16_t lAf[2][8 * 512];   // 8 A tiles / buffer
    __shared__ __align__(16) bf16_t lBf[2][4 * 512];   // 4 B tiles / buffer

    const int tid  = threadIdx.x;
    const int wave = tid >> 5;
    const int lane = tid & 31;
    const int mBlk = blockIdx.y * 128;
    const int nBlk = blockIdx.x * 64;
    const int mW   = (wave >> 1) * 32;
    const int nW   = (wave & 1) * 32;
    const int tM   = (wave >> 1) * 2;    // first A fragment tile
    const int tN   = (wave & 1) * 2;     // first B fragment tile
    const int kb   = lane >> 4;
    const int l16  = lane & 15;

    // staging coordinates (all 256 threads)
    const int ar = tid >> 1;                 // A row 0..127
    const int ac = (tid & 1) * 16;           // A col base {0,16}
    const int bn = tid >> 2;                 // B row (n) 0..63
    const int bc = (tid & 3) * 8;            // B col (k) {0,8,16,24}

    v8f acc00 = {}, acc01 = {}, acc10 = {}, acc11 = {};

    auto stage = [&](int k0, int buf) {
        // ---- A tile 128x32: 2 x 16B async copies per thread ----
        const bf16_t* ga = A + (size_t)(mBlk + ar) * lda + (k0 + ac);
        const int atile = ar >> 4;
        #pragma unroll
        for (int j = 0; j < 2; ++j) {
            const int c = ac + j * 8;
            const int ln = (ar & 15) + 16 * ((c >> 3) & 1);
            const int eo = (c & 16) ? 8 : 0;
            unsigned lds = (unsigned)(uintptr_t)&lAf[buf][atile * 512 + ln * 16 + eo];
            async_ld_b128(lds, ga + j * 8);
        }
        // ---- B tile 64x32: 1 x 16B async copy per thread ----
        const bf16_t* gb = Bt + (size_t)(nBlk + bn) * ldb + (k0 + bc);
        const int btile = bn >> 4;
        const int ln = (bn & 15) + 16 * ((bc >> 3) & 1);
        const int eo = (bc & 16) ? 8 : 0;
        unsigned lds = (unsigned)(uintptr_t)&lBf[buf][btile * 512 + ln * 16 + eo];
        async_ld_b128(lds, gb);
    };

    stage(0, 0);
    wait_async0();
    __syncthreads();

    const int KT = K >> 5;
    for (int kt = 0; kt < KT; ++kt) {
        const int buf = kt & 1;
        if (kt + 1 < KT) stage((kt + 1) << 5, buf ^ 1);   // overlap next tile

        const v16bf a0 = *(const v16bf*)&lAf[buf][(tM + 0) * 512 + lane * 16];
        const v16bf a1 = *(const v16bf*)&lAf[buf][(tM + 1) * 512 + lane * 16];
        const v16bf b0 = *(const v16bf*)&lBf[buf][(tN + 0) * 512 + lane * 16];
        const v16bf b1 = *(const v16bf*)&lBf[buf][(tN + 1) * 512 + lane * 16];

        acc00 = __builtin_amdgcn_wmma_f32_16x16x32_bf16(false, a0, false, b0, (short)0, acc00, false, false);
        acc01 = __builtin_amdgcn_wmma_f32_16x16x32_bf16(false, a0, false, b1, (short)0, acc01, false, false);
        acc10 = __builtin_amdgcn_wmma_f32_16x16x32_bf16(false, a1, false, b0, (short)0, acc10, false, false);
        acc11 = __builtin_amdgcn_wmma_f32_16x16x32_bf16(false, a1, false, b1, (short)0, acc11, false, false);

        wait_async0();
        __syncthreads();
    }

    // ---- fused epilogue; C/D layout: VGPR r -> M = r + 8*kb, N = l16 ----
    #pragma unroll
    for (int mi = 0; mi < 2; ++mi) {
        #pragma unroll
        for (int ni = 0; ni < 2; ++ni) {
            v8f acc = (mi == 0) ? (ni == 0 ? acc00 : acc01)
                                : (ni == 0 ? acc10 : acc11);
            const int cn = nBlk + nW + ni * 16 + l16;
            #pragma unroll
            for (int r = 0; r < 8; ++r) {
                const int cm = mBlk + mW + mi * 16 + r + 8 * kb;
                float v = acc[r];
                if (bias) v += bias[cn];
                if (act == ACT_RELU)      v = fmaxf(v, 0.0f);
                else if (act == ACT_GELU) v = 0.5f * v * (1.0f + erff(v * 0.70710678118654752f));
                if (residual) v += residual[(size_t)cm * ldres + cn];
                if (Cf)  Cf[(size_t)cm * ldc + cn] = v;
                if (Cb)  Cb[(size_t)cm * ldc + cn] = (bf16_t)v;
                if (CbT) CbT[(size_t)cn * ldcT + cm] = (bf16_t)v;
            }
        }
    }
}

// ---------------- helpers ----------------
// transpose-cast: src [R,C] f32 -> dst [C,R] bf16
__global__ void transp_cast(const float* __restrict__ s, bf16_t* __restrict__ d, int R, int C)
{
    size_t idx = (size_t)blockIdx.x * blockDim.x + threadIdx.x;
    if (idx >= (size_t)R * C) return;
    int r = (int)(idx / C), c = (int)(idx % C);
    d[(size_t)c * R + r] = (bf16_t)s[idx];
}

// im2col for Conv1d(k=3, pad=1): xcol[(b*T+t), k*DFEAT+c] = x[b, t+k-1, c]
__global__ void im2col_k3(const float* __restrict__ x, bf16_t* __restrict__ xc)
{
    size_t idx = (size_t)blockIdx.x * blockDim.x + threadIdx.x;
    const size_t total = (size_t)NB * NT * 3 * DFEAT;
    if (idx >= total) return;
    int col = (int)(idx % (3 * DFEAT));
    int row = (int)(idx / (3 * DFEAT));
    int b = row / NT, t = row % NT;
    int k = col / DFEAT, c = col % DFEAT;
    int ts = t + k - 1;
    float v = (ts >= 0 && ts < NT) ? x[((size_t)b * NT + ts) * DFEAT + c] : 0.0f;
    xc[idx] = (bf16_t)v;
}

// emb_w [DIM, DFEAT, 3] (OIW) -> WcT [DIM, 3*DFEAT] bf16 (transposed form)
__global__ void pack_embw_T(const float* __restrict__ w, bf16_t* __restrict__ wc)
{
    size_t idx = (size_t)blockIdx.x * blockDim.x + threadIdx.x;
    const size_t total = (size_t)NDIM * 3 * DFEAT;
    if (idx >= total) return;
    int o = (int)(idx / (3 * DFEAT));
    int row = (int)(idx % (3 * DFEAT));
    int k = row / DFEAT, c = row % DFEAT;
    wc[idx] = (bf16_t)w[((size_t)o * DFEAT + c) * 3 + k];
}

__global__ void layernorm_rows(const float* __restrict__ x, const float* __restrict__ g,
                               const float* __restrict__ b, bf16_t* __restrict__ z, int n)
{
    __shared__ float sm[256];
    const int row = blockIdx.x, tid = threadIdx.x;
    const float* xr = x + (size_t)row * n;
    float s = 0.0f;
    for (int c = tid; c < n; c += 256) s += xr[c];
    sm[tid] = s; __syncthreads();
    for (int st = 128; st > 0; st >>= 1) { if (tid < st) sm[tid] += sm[tid + st]; __syncthreads(); }
    const float mean = sm[0] / n; __syncthreads();
    float v = 0.0f;
    for (int c = tid; c < n; c += 256) { float d = xr[c] - mean; v += d * d; }
    sm[tid] = v; __syncthreads();
    for (int st = 128; st > 0; st >>= 1) { if (tid < st) sm[tid] += sm[tid + st]; __syncthreads(); }
    const float inv = rsqrtf(sm[0] / n + EPSF);
    for (int c = tid; c < n; c += 256)
        z[(size_t)row * n + c] = (bf16_t)((xr[c] - mean) * inv * g[c] + b[c]);
}

// fused scale + row softmax: f32 logits in (overwritten with exp), bf16 probs out
__global__ void softmax_rows(float* __restrict__ s, bf16_t* __restrict__ p, int n, float scale)
{
    __shared__ float sm[256];
    const int row = blockIdx.x, tid = threadIdx.x;
    float* sr = s + (size_t)row * n;
    bf16_t* pr = p + (size_t)row * n;
    float mx = -3.4e38f;
    for (int c = tid; c < n; c += 256) mx = fmaxf(mx, sr[c] * scale);
    sm[tid] = mx; __syncthreads();
    for (int st = 128; st > 0; st >>= 1) { if (tid < st) sm[tid] = fmaxf(sm[tid], sm[tid + st]); __syncthreads(); }
    mx = sm[0]; __syncthreads();
    float sum = 0.0f;
    for (int c = tid; c < n; c += 256) { float e = expf(sr[c] * scale - mx); sr[c] = e; sum += e; }
    sm[tid] = sum; __syncthreads();
    for (int st = 128; st > 0; st >>= 1) { if (tid < st) sm[tid] += sm[tid + st]; __syncthreads(); }
    const float inv = 1.0f / sm[0];
    for (int c = tid; c < n; c += 256) pr[c] = (bf16_t)(sr[c] * inv);
}

// attn2[i,j] = exp(-|i-j|/e) / colsum(j)
__global__ void a2_colsum(float* __restrict__ cs, int n)
{
    __shared__ float sm[256];
    const int j = blockIdx.x, tid = threadIdx.x;
    float s = 0.0f;
    for (int i = tid; i < n; i += 256) s += expf(-fabsf((float)(i - j)) * INV_E);
    sm[tid] = s; __syncthreads();
    for (int st = 128; st > 0; st >>= 1) { if (tid < st) sm[tid] += sm[tid + st]; __syncthreads(); }
    if (tid == 0) cs[j] = sm[0];
}
__global__ void a2_fill(const float* __restrict__ cs, bf16_t* __restrict__ a2, int n)
{
    const int i = blockIdx.x, tid = threadIdx.x;
    for (int j = tid; j < n; j += 256)
        a2[(size_t)i * n + j] = (bf16_t)(expf(-fabsf((float)(i - j)) * INV_E) / cs[j]);
}

// NormalHead: per (b,t): 512->32 (BN+ReLU) ->16 (BN+ReLU) ->1, sigmoid
__global__ void head_kernel(const float* __restrict__ h,
                            const float* __restrict__ h1w, const float* __restrict__ h1b,
                            const float* __restrict__ g1, const float* __restrict__ b1,
                            const float* __restrict__ m1, const float* __restrict__ v1,
                            const float* __restrict__ h2w, const float* __restrict__ h2b,
                            const float* __restrict__ g2, const float* __restrict__ b2,
                            const float* __restrict__ m2, const float* __restrict__ v2,
                            const float* __restrict__ h3w, const float* __restrict__ h3b,
                            float* __restrict__ out)
{
    const int idx = blockIdx.x * blockDim.x + threadIdx.x;   // b*T + t
    if (idx >= NB * NT) return;
    const float* hr = h + (size_t)idx * NDIM;
    float f1[NRED1];
    #pragma unroll 1
    for (int o = 0; o < NRED1; ++o) {
        float s = h1b[o];
        #pragma unroll 4
        for (int c = 0; c < NDIM; ++c) s += h1w[(size_t)o * NDIM + c] * hr[c];
        s = (s - m1[o]) * rsqrtf(v1[o] + EPSF) * g1[o] + b1[o];
        f1[o] = fmaxf(s, 0.0f);
    }
    float f2[NRED2];
    #pragma unroll 1
    for (int o = 0; o < NRED2; ++o) {
        float s = h2b[o];
        #pragma unroll
        for (int c = 0; c < NRED1; ++c) s += h2w[o * NRED1 + c] * f1[c];
        s = (s - m2[o]) * rsqrtf(v2[o] + EPSF) * g2[o] + b2[o];
        f2[o] = fmaxf(s, 0.0f);
    }
    float f3 = h3b[0];
    #pragma unroll
    for (int c = 0; c < NRED2; ++c) f3 += h3w[c] * f2[c];
    out[idx] = 1.0f / (1.0f + expf(-f3));
}

// =====================================================================
extern "C" void kernel_launch(void* const* d_in, const int* in_sizes, int n_in,
                              void* d_out, int out_size, void* d_ws, size_t ws_size,
                              hipStream_t stream)
{
    const float* x      = (const float*)d_in[0];
    const float* emb_w  = (const float*)d_in[1];
    const float* emb_b  = (const float*)d_in[2];
    const float* ln1_g  = (const float*)d_in[3];
    const float* ln1_b  = (const float*)d_in[4];
    const float* wqkvt  = (const float*)d_in[5];
    const float* wout   = (const float*)d_in[6];
    const float* bout   = (const float*)d_in[7];
    const float* ln2_g  = (const float*)d_in[8];
    const float* ln2_b  = (const float*)d_in[9];
    const float* w1     = (const float*)d_in[10];
    const float* b1     = (const float*)d_in[11];
    const float* w2     = (const float*)d_in[12];
    const float* b2     = (const float*)d_in[13];
    const float* h1w    = (const float*)d_in[14];
    const float* h1b    = (const float*)d_in[15];
    const float* bn1_g  = (const float*)d_in[16];
    const float* bn1_b  = (const float*)d_in[17];
    const float* bn1_m  = (const float*)d_in[18];
    const float* bn1_v  = (const float*)d_in[19];
    const float* h2w    = (const float*)d_in[20];
    const float* h2b    = (const float*)d_in[21];
    const float* bn2_g  = (const float*)d_in[22];
    const float* bn2_b  = (const float*)d_in[23];
    const float* bn2_m  = (const float*)d_in[24];
    const float* bn2_v  = (const float*)d_in[25];
    const float* h3w    = (const float*)d_in[26];
    const float* h3b    = (const float*)d_in[27];
    float* out = (float*)d_out;
    (void)in_sizes; (void)n_in; (void)out_size; (void)ws_size;

    // ---- bump allocator over d_ws (~125 MB) ----
    char* wsB = (char*)d_ws;
    size_t off = 0;
    auto alloc = [&](size_t bytes) -> void* {
        off = (off + 255) & ~(size_t)255;
        void* p = wsB + off; off += bytes; return p;
    };
    const int ROWS = NB * NT;   // 4096
    bf16_t* xcol    = (bf16_t*)alloc((size_t)ROWS * 3 * DFEAT * 2);
    bf16_t* wcembT  = (bf16_t*)alloc((size_t)NDIM * 3 * DFEAT * 2);         // [512, 3072]
    bf16_t* wqkvtT  = (bf16_t*)alloc((size_t)2 * 4 * NINNER * NDIM * 2);    // [l][2048, 512]
    bf16_t* woutT   = (bf16_t*)alloc((size_t)2 * NDIM * 2 * NINNER * 2);    // [l][512, 1024]
    bf16_t* w1T     = (bf16_t*)alloc((size_t)2 * NMLP * NDIM * 2);          // [l][512, 512]
    bf16_t* w2T     = (bf16_t*)alloc((size_t)2 * NDIM * NMLP * 2);          // [l][512, 512]
    float*  hbuf    = (float*) alloc((size_t)ROWS * NDIM * 4);
    bf16_t* zb      = (bf16_t*)alloc((size_t)ROWS * NDIM * 2);
    bf16_t* qkvtb   = (bf16_t*)alloc((size_t)ROWS * 4 * NINNER * 2);        // [4096, 2048]
    bf16_t* qkvtT   = (bf16_t*)alloc((size_t)4 * NINNER * ROWS * 2);        // [2048, 4096]
    bf16_t* a2b     = (bf16_t*)alloc((size_t)NT * NT * 2);
    float*  a2cs    = (float*) alloc((size_t)NT * 4);
    float*  scores  = (float*) alloc((size_t)NT * NT * 4);
    bf16_t* probs   = (bf16_t*)alloc((size_t)NT * NT * 2);
    bf16_t* attcat  = (bf16_t*)alloc((size_t)ROWS * 2 * NINNER * 2);
    bf16_t* midb    = (bf16_t*)alloc((size_t)ROWS * NMLP * 2);

    auto gemm = [&](const bf16_t* A, int lda, const bf16_t* Bt, int ldb,
                    float* Cf, bf16_t* Cb, bf16_t* CbT, int ldc, int ldcT,
                    const float* bias, const float* res, int ldres,
                    int M, int N, int K, int act) {
        dim3 g(N / 64, M / 128);
        gemm_bf16_wmma<<<g, 256, 0, stream>>>(A, lda, Bt, ldb, Cf, Cb, CbT,
                                              ldc, ldcT, bias, res, ldres, K, act);
    };
    auto tcast = [&](const float* s, bf16_t* d, int R, int C) {
        size_t n = (size_t)R * C;
        transp_cast<<<(int)((n + 255) / 256), 256, 0, stream>>>(s, d, R, C);
    };

    // ---- one-time prep: transposed bf16 weights, im2col, attn2 ----
    for (int l = 0; l < 2; ++l) {
        tcast(wqkvt + (size_t)l * NDIM * 4 * NINNER, wqkvtT + (size_t)l * 4 * NINNER * NDIM, NDIM, 4 * NINNER);
        tcast(wout  + (size_t)l * 2 * NINNER * NDIM, woutT  + (size_t)l * NDIM * 2 * NINNER, 2 * NINNER, NDIM);
        tcast(w1    + (size_t)l * NDIM * NMLP,       w1T    + (size_t)l * NMLP * NDIM,       NDIM, NMLP);
        tcast(w2    + (size_t)l * NMLP * NDIM,       w2T    + (size_t)l * NDIM * NMLP,       NMLP, NDIM);
    }
    {
        size_t n = (size_t)NDIM * 3 * DFEAT;
        pack_embw_T<<<(int)((n + 255) / 256), 256, 0, stream>>>(emb_w, wcembT);
        size_t m = (size_t)ROWS * 3 * DFEAT;
        im2col_k3<<<(int)((m + 255) / 256), 256, 0, stream>>>(x, xcol);
    }
    a2_colsum<<<NT, 256, 0, stream>>>(a2cs, NT);
    a2_fill  <<<NT, 256, 0, stream>>>(a2cs, a2b, NT);

    // ---- embedding: h = relu(xcol @ Wc + emb_b) ----
    gemm(xcol, 3 * DFEAT, wcembT, 3 * DFEAT, hbuf, nullptr, nullptr, NDIM, 0,
         emb_b, nullptr, 0, ROWS, NDIM, 3 * DFEAT, ACT_RELU);

    const float scale = 0.125f;   // DH^-0.5
    for (int l = 0; l < 2; ++l) {
        // LN1 -> zb (bf16)
        layernorm_rows<<<ROWS, 256, 0, stream>>>(hbuf, ln1_g + l * NDIM, ln1_b + l * NDIM, zb, NDIM);
        // qkvt = zb @ Wqkvt[l] : write row-major AND transposed copies
        gemm(zb, NDIM, wqkvtT + (size_t)l * 4 * NINNER * NDIM, NDIM,
             nullptr, qkvtb, qkvtT, 4 * NINNER, ROWS,
             nullptr, nullptr, 0, ROWS, 4 * NINNER, NDIM, ACT_NONE);

        for (int b = 0; b < NB; ++b) {
            const bf16_t* qkvt_b = qkvtb + (size_t)b * NT * 4 * NINNER;
            for (int hh = 0; hh < NH; ++hh) {
                const bf16_t* Q  = qkvt_b + 0 * NINNER + hh * NDH;                 // [T,64] ld 2048
                const bf16_t* Kt = qkvt_b + 1 * NINNER + hh * NDH;                 // [T,64] = Bt for QK^T
                const bf16_t* Vt = qkvtT + (size_t)(2 * NINNER + hh * NDH) * ROWS + b * NT; // [64,T] ld 4096
                const bf16_t* Tt = qkvtT + (size_t)(3 * NINNER + hh * NDH) * ROWS + b * NT; // [64,T] ld 4096
                bf16_t* o1 = attcat + (size_t)b * NT * 2 * NINNER + hh * 2 * NDH;
                bf16_t* o2 = o1 + NDH;
                // dots = Q @ K^T
                gemm(Q, 4 * NINNER, Kt, 4 * NINNER, scores, nullptr, nullptr, NT, 0,
                     nullptr, nullptr, 0, NT, NT, NDH, ACT_NONE);
                // probs = softmax(dots * scale)
                softmax_rows<<<NT, 256, 0, stream>>>(scores, probs, NT, scale);
                // out1 = probs @ V  -> attcat[:, h*128 + 0..63]
                gemm(probs, NT, Vt, ROWS, nullptr, o1, nullptr, 2 * NINNER, 0,
                     nullptr, nullptr, 0, NT, NDH, NT, ACT_NONE);
                // out2 = attn2 @ Tm -> attcat[:, h*128 + 64..127]
                gemm(a2b, NT, Tt, ROWS, nullptr, o2, nullptr, 2 * NINNER, 0,
                     nullptr, nullptr, 0, NT, NDH, NT, ACT_NONE);
            }
        }
        // h += attcat @ Wout[l] + bout[l]
        gemm(attcat, 2 * NINNER, woutT + (size_t)l * NDIM * 2 * NINNER, 2 * NINNER,
             hbuf, nullptr, nullptr, NDIM, 0, bout + l * NDIM, hbuf, NDIM,
             ROWS, NDIM, 2 * NINNER, ACT_NONE);
        // MLP: mid = gelu(LN2(h) @ W1 + b1); h += mid @ W2 + b2
        layernorm_rows<<<ROWS, 256, 0, stream>>>(hbuf, ln2_g + l * NDIM, ln2_b + l * NDIM, zb, NDIM);
        gemm(zb, NDIM, w1T + (size_t)l * NMLP * NDIM, NDIM,
             nullptr, midb, nullptr, NMLP, 0, b1 + l * NMLP, nullptr, 0,
             ROWS, NMLP, NDIM, ACT_GELU);
        gemm(midb, NMLP, w2T + (size_t)l * NDIM * NMLP, NMLP,
             hbuf, nullptr, nullptr, NDIM, 0, b2 + l * NDIM, hbuf, NDIM,
             ROWS, NDIM, NMLP, ACT_NONE);
    }

    // ---- head ----
    head_kernel<<<(ROWS + 255) / 256, 256, 0, stream>>>(
        hbuf, h1w, h1b, bn1_g, bn1_b, bn1_m, bn1_v,
        h2w, h2b, bn2_g, bn2_b, bn2_m, bn2_v, h3w, h3b, out);
}
